// SpatialAttentionBlock3d_89670327206518
// MI455X (gfx1250) — matrix operand order repaired and checked by reading, not compiled, not run on hardware
//
#include <hip/hip_runtime.h>

typedef __attribute__((ext_vector_type(16))) _Float16 v16h;
typedef __attribute__((ext_vector_type(8)))  _Float16 v8h;
typedef __attribute__((ext_vector_type(8)))  float    v8f;

#define BSZ  2
#define CCH  256
#define NPOS 4096
#define KTOT 768   // 3 taps * 256 cin

#define KROW 264   // padded LDS row stride for K tile (bank-conflict-free)
#define VROW 40    // padded LDS row stride for V tile

// CDNA5 async global->LDS path (ASYNCcnt-tracked), with safe fallback.
#if defined(__has_builtin)
#  if __has_builtin(__builtin_amdgcn_global_load_async_to_lds_b128) && \
      __has_builtin(__builtin_amdgcn_s_wait_asynccnt)
#    define USE_ASYNC_LDS 1
#  endif
#endif
#ifndef USE_ASYNC_LDS
#  define USE_ASYNC_LDS 0
#endif

// The b128 async builtin takes pointers to <4 x i32> chunks (per compiler
// diagnostic): arg0 = global (AS1) source, arg1 = LDS (AS3) destination.
typedef int v4i __attribute__((vector_size(16)));
typedef __attribute__((address_space(1))) v4i gv4i;
typedef __attribute__((address_space(3))) v4i lv4i;

__device__ __forceinline__ void cp16(const _Float16* g, _Float16* l) {
#if USE_ASYNC_LDS
  __builtin_amdgcn_global_load_async_to_lds_b128((gv4i*)g, (lv4i*)l, 0, 0);
#else
  *(v8h*)l = *(const v8h*)g;
#endif
}

__device__ __forceinline__ v8f wmma16(v16h a, v16h b, v8f c) {
  return __builtin_amdgcn_wmma_f32_16x16x32_f16(false, a, false, b, (short)0, c, false, false);
}

__device__ __forceinline__ v16h cat8(v8h lo, v8h hi) {
  return __builtin_shufflevector(lo, hi, 0,1,2,3,4,5,6,7,8,9,10,11,12,13,14,15);
}

// ---------------------------------------------------------------------------
// 1) Repack QKV weights: f32 [cout][cin][tap] -> f16 [proj][cout][tap*256+cin]
// ---------------------------------------------------------------------------
__global__ void k_prep_w(const float* __restrict__ wq, const float* __restrict__ wk,
                         const float* __restrict__ wv, _Float16* __restrict__ Wr) {
  int i = blockIdx.x * 256 + threadIdx.x;            // 3*256*768
  if (i >= 3 * CCH * KTOT) return;
  int proj = i / (CCH * KTOT);
  int rem  = i % (CCH * KTOT);
  int cout = rem / KTOT;
  int k    = rem % KTOT;
  int tap  = k / CCH;
  int cin  = k % CCH;
  const float* w = (proj == 0) ? wq : (proj == 1) ? wk : wv;
  Wr[i] = (_Float16)w[(cout * CCH + cin) * 3 + tap];
}

// ---------------------------------------------------------------------------
// 2) Channel-split 1x1 convs + SiLU. Writes xg f32 [b][c][n] (residual) and
//    xg f16 position-major [b][n][c] (GEMM operand).
// ---------------------------------------------------------------------------
__global__ void k_split_gate(const float* __restrict__ x,
                             const float* __restrict__ w1, const float* __restrict__ b1,
                             const float* __restrict__ w2, const float* __restrict__ b2,
                             const float* __restrict__ w3, const float* __restrict__ b3,
                             const float* __restrict__ w4, const float* __restrict__ b4,
                             float* __restrict__ xg, _Float16* __restrict__ xgt) {
  int i = blockIdx.x * 256 + threadIdx.x;            // B*C*N
  if (i >= BSZ * CCH * NPOS) return;
  int n = i % NPOS;
  int c = (i / NPOS) % CCH;
  int b = i / (NPOS * CCH);
  int g  = c >> 6;
  int co = c & 63;
  const float* wg = (g == 0) ? w1 : (g == 1) ? w2 : (g == 2) ? w3 : w4;
  const float* bg = (g == 0) ? b1 : (g == 1) ? b2 : (g == 2) ? b3 : b4;
  const float* xin = x + ((size_t)(b * CCH + g * 64)) * NPOS + n;
  const float* wr  = wg + co * 64;
  float acc = bg[co];
  #pragma unroll 8
  for (int ci = 0; ci < 64; ++ci) acc += wr[ci] * xin[(size_t)ci * NPOS];
  float sgate = acc / (1.0f + __expf(-acc));
  xg[i] = sgate;
  xgt[((size_t)(b * NPOS) + n) * CCH + c] = (_Float16)sgate;
}

// ---------------------------------------------------------------------------
// 3) QKV directional 3-tap projections via WMMA.
//    grid = (N/16, C/16, 3*B), block = 32 (one wave per 16x16 tile).
//    raw layout: [proj][b][cout][pos] f32 (bias added later in LN).
// ---------------------------------------------------------------------------
__global__ __launch_bounds__(32)
void k_qkv_gemm(const _Float16* __restrict__ xgt, const _Float16* __restrict__ Wr,
                float* __restrict__ raw) {
  const int lane = threadIdx.x & 31;
  const int ln   = lane & 15;     // N (pos) within tile / M (cout) for A operand
  const int s    = lane >> 4;     // lane half
  const int ptile = blockIdx.x;
  const int ctile = blockIdx.y;
  const int pb    = blockIdx.z;   // proj*BSZ + b
  const int proj  = pb >> 1;
  const int b     = pb & 1;

  const int pos = ptile * 16 + ln;
  const int d = pos >> 8, h = (pos >> 4) & 15, w = pos & 15;

  int  spos[3];
  bool val[3];
  #pragma unroll
  for (int t = 0; t < 3; ++t) {
    int dl = t - 1;
    if (proj == 0)      { int hh = h + dl; val[t] = (hh >= 0 && hh < 16); spos[t] = pos + dl * 16;  }
    else if (proj == 1) { int ww = w + dl; val[t] = (ww >= 0 && ww < 16); spos[t] = pos + dl;       }
    else                { int dd = d + dl; val[t] = (dd >= 0 && dd < 16); spos[t] = pos + dl * 256; }
  }

  const _Float16* wrow = Wr + (size_t)proj * CCH * KTOT + (size_t)(ctile * 16 + ln) * KTOT;
  v8f acc = {};

  for (int tap = 0; tap < 3; ++tap) {
    const _Float16* xb = val[tap]
        ? (xgt + ((size_t)(b * NPOS) + spos[tap]) * CCH) : (const _Float16*)0;
    #pragma unroll
    for (int kc = 0; kc < CCH; kc += 32) {
      int k0 = tap * CCH + kc;
      v8h alo = *(const v8h*)(wrow + k0 + s * 8);
      v8h ahi = *(const v8h*)(wrow + k0 + 16 + s * 8);
      v16h am = cat8(alo, ahi);
      v16h bm = {};
      if (xb) bm = cat8(*(const v8h*)(xb + kc + s * 8), *(const v8h*)(xb + kc + 16 + s * 8));
      acc = wmma16(am, bm, acc);
    }
  }

  float* out = raw + (size_t)pb * CCH * NPOS;
  #pragma unroll
  for (int r = 0; r < 8; ++r) {
    int cout = ctile * 16 + s * 8 + r;
    out[(size_t)cout * NPOS + ptile * 16 + ln] = acc[r];
  }
}

// ---------------------------------------------------------------------------
// 4) Channel LayerNorm per position. Q -> position-major f16 (x 1/16 score
//    scale), K -> position-major f16, V -> channel-major f16.
// ---------------------------------------------------------------------------
__global__ void k_ln(const float* __restrict__ raw,
                     const float* __restrict__ bq, const float* __restrict__ bk,
                     const float* __restrict__ bv,
                     const float* __restrict__ qs, const float* __restrict__ qb,
                     const float* __restrict__ ks, const float* __restrict__ kb,
                     const float* __restrict__ vs, const float* __restrict__ vb,
                     _Float16* __restrict__ Qt, _Float16* __restrict__ Kt,
                     _Float16* __restrict__ Vc) {
  int i = blockIdx.x * 256 + threadIdx.x;            // 3*B*N
  if (i >= 3 * BSZ * NPOS) return;
  int pos  = i % NPOS;
  int b    = (i / NPOS) % BSZ;
  int proj = i / (NPOS * BSZ);
  const float* bias = (proj == 0) ? bq : (proj == 1) ? bk : bv;
  const float* sc   = (proj == 0) ? qs : (proj == 1) ? ks : vs;
  const float* sh   = (proj == 0) ? qb : (proj == 1) ? kb : vb;
  const float* r = raw + (size_t)(proj * BSZ + b) * CCH * NPOS + pos;

  float sum = 0.0f, sq = 0.0f;
  for (int c = 0; c < CCH; ++c) {
    float v = r[(size_t)c * NPOS] + bias[c];
    sum += v; sq += v * v;
  }
  float mu   = sum * (1.0f / CCH);
  float var  = sq * (1.0f / CCH) - mu * mu;
  float rstd = rsqrtf(var + 1e-5f);

  for (int c = 0; c < CCH; ++c) {
    float v = (r[(size_t)c * NPOS] + bias[c] - mu) * rstd * sc[c] + sh[c];
    if (proj == 0)      Qt[((size_t)(b * NPOS) + pos) * CCH + c] = (_Float16)(v * 0.0625f);
    else if (proj == 1) Kt[((size_t)(b * NPOS) + pos) * CCH + c] = (_Float16)v;
    else                Vc[((size_t)(b * CCH) + c) * NPOS + pos] = (_Float16)v;
  }
}

// ---------------------------------------------------------------------------
// 5) Flash attention + residual epilogue. 8 waves/block, one 16-query tile
//    per wave. K/V tiles staged cooperatively into LDS once per block
//    (8x L2-traffic reduction) via CDNA5 async global->LDS when available.
// ---------------------------------------------------------------------------
__global__ __launch_bounds__(256)
void k_attn(const _Float16* __restrict__ Qt, const _Float16* __restrict__ Kt,
            const _Float16* __restrict__ Vc, const float* __restrict__ xg,
            const float* __restrict__ gamma, float* __restrict__ out) {
  __shared__ _Float16 Ks[32 * KROW];    // 32 keys x 256 ch (padded rows)
  __shared__ _Float16 Vs[CCH * VROW];   // 256 ch x 32 keys (padded rows)
  __shared__ _Float16 Pl[8 * 16 * 32];  // per-wave P transpose buffers

  const int tid  = threadIdx.x;
  const int lane = tid & 31;
  const int wid  = tid >> 5;
  const int ln   = lane & 15;
  const int s    = lane >> 4;
  const int tile = blockIdx.x * 8 + wid;   // 0..511 ; b uniform per block
  const int b    = tile >> 8;
  const int qt   = tile & 255;
  _Float16* Pw = &Pl[wid * 16 * 32];

  // Stage Q tile as 8 A-operands (16 queries x 256 channels, pre-scaled 1/16)
  v16h aq[8];
  {
    const _Float16* qrow = Qt + ((size_t)(b * NPOS) + qt * 16 + ln) * CCH;
    #pragma unroll
    for (int kc8 = 0; kc8 < 8; ++kc8) {
      int kc = kc8 * 32;
      aq[kc8] = cat8(*(const v8h*)(qrow + kc + s * 8),
                     *(const v8h*)(qrow + kc + 16 + s * 8));
    }
  }

  v8f o[16];
  v8f vz = {};
  #pragma unroll
  for (int t = 0; t < 16; ++t) o[t] = vz;
  float m[8], l[8];
  #pragma unroll
  for (int r = 0; r < 8; ++r) { m[r] = -1e30f; l[r] = 0.0f; }

  const _Float16* kb_g = Kt + (size_t)b * NPOS * CCH;
  const _Float16* vb_g = Vc + (size_t)b * CCH * NPOS;

  for (int kt = 0; kt < NPOS; kt += 32) {
    __syncthreads();   // all waves finished consuming previous K/V tiles

    // ---- cooperative staging: K tile (16 KB contiguous) + V tile (256x64B)
    {
      const _Float16* ksrc = kb_g + (size_t)kt * CCH;
      #pragma unroll
      for (int j = 0; j < 4; ++j) {
        int c8 = tid + j * 256;                   // 16-byte chunk id, 0..1023
        cp16(ksrc + c8 * 8, &Ks[(c8 >> 5) * KROW + (c8 & 31) * 8]);
      }
      const _Float16* vsrc = vb_g + (size_t)tid * NPOS + kt;
      #pragma unroll
      for (int j = 0; j < 4; ++j)
        cp16(vsrc + j * 8, &Vs[tid * VROW + j * 8]);
      if (kt + 32 < NPOS)
        __builtin_prefetch(kb_g + (size_t)(kt + 32) * CCH + tid * 32, 0, 1);
    }
#if USE_ASYNC_LDS
    __builtin_amdgcn_s_wait_asynccnt(0);
#endif
    __syncthreads();   // staged tiles visible to all waves

    // ---- S = Q @ K^T (two 16-key subtiles), operands from LDS
    v8f s0 = {}, s1 = {};
    #pragma unroll
    for (int kc8 = 0; kc8 < 8; ++kc8) {
      const _Float16* r0 = &Ks[ln * KROW + kc8 * 32 + s * 8];
      const _Float16* r1 = &Ks[(16 + ln) * KROW + kc8 * 32 + s * 8];
      v16h b0 = cat8(*(const v8h*)(r0), *(const v8h*)(r0 + 16));
      v16h b1 = cat8(*(const v8h*)(r1), *(const v8h*)(r1 + 16));
      s0 = wmma16(aq[kc8], b0, s0);
      s1 = wmma16(aq[kc8], b1, s1);
    }

    // ---- online softmax per query row (row s*8+r lives in this lane half)
    float alpha[8];
    #pragma unroll
    for (int r = 0; r < 8; ++r) {
      float t0 = fmaxf(s0[r], s1[r]);
      #pragma unroll
      for (int off = 1; off < 16; off <<= 1) t0 = fmaxf(t0, __shfl_xor(t0, off, 32));
      float mn = fmaxf(m[r], t0);
      float p0 = __expf(s0[r] - mn);
      float p1 = __expf(s1[r] - mn);
      int row = s * 8 + r;
      Pw[row * 32 + ln]      = (_Float16)p0;
      Pw[row * 32 + 16 + ln] = (_Float16)p1;
      float ps = p0 + p1;
      #pragma unroll
      for (int off = 1; off < 16; off <<= 1) ps += __shfl_xor(ps, off, 32);
      alpha[r] = __expf(m[r] - mn);
      l[r] = l[r] * alpha[r] + ps;
      m[r] = mn;
    }

    // ---- transpose P (C-layout -> A-layout) via per-wave LDS region.
    // LDS ops are in-order within a wave; no block barrier needed here.
    v16h pa = cat8(*(const v8h*)(&Pw[ln * 32 + s * 8]),
                   *(const v8h*)(&Pw[ln * 32 + 16 + s * 8]));

    // ---- rescale O and accumulate P @ V over all 16 channel subtiles
    #pragma unroll
    for (int t = 0; t < 16; ++t) {
      #pragma unroll
      for (int r = 0; r < 8; ++r) o[t][r] *= alpha[r];
      const _Float16* vr = &Vs[(t * 16 + ln) * VROW + s * 8];
      v16h bvv = cat8(*(const v8h*)(vr), *(const v8h*)(vr + 16));
      o[t] = wmma16(pa, bvv, o[t]);
    }
  }

  // ---- epilogue: out[b][c][n] = gamma * (O/l) + xg[b][c][n]
  float g = gamma[0];
  const float* xgb = xg + (size_t)b * CCH * NPOS;
  float* ob = out + (size_t)b * CCH * NPOS;
  #pragma unroll
  for (int t = 0; t < 16; ++t) {
    #pragma unroll
    for (int r = 0; r < 8; ++r) {
      int ch  = t * 16 + ln;
      int row = qt * 16 + s * 8 + r;
      size_t idx = (size_t)ch * NPOS + row;
      ob[idx] = g * (o[t][r] / l[r]) + xgb[idx];
    }
  }
}

// ---------------------------------------------------------------------------
extern "C" void kernel_launch(void* const* d_in, const int* in_sizes, int n_in,
                              void* d_out, int out_size, void* d_ws, size_t ws_size,
                              hipStream_t stream) {
  const float* x  = (const float*)d_in[0];
  const float* w1 = (const float*)d_in[1];  const float* b1 = (const float*)d_in[2];
  const float* w2 = (const float*)d_in[3];  const float* b2 = (const float*)d_in[4];
  const float* w3 = (const float*)d_in[5];  const float* b3 = (const float*)d_in[6];
  const float* w4 = (const float*)d_in[7];  const float* b4 = (const float*)d_in[8];
  const float* wq = (const float*)d_in[9];  const float* bq = (const float*)d_in[10];
  const float* wk = (const float*)d_in[11]; const float* bk = (const float*)d_in[12];
  const float* wv = (const float*)d_in[13]; const float* bv = (const float*)d_in[14];
  const float* qs = (const float*)d_in[15]; const float* qb = (const float*)d_in[16];
  const float* ks = (const float*)d_in[17]; const float* kb = (const float*)d_in[18];
  const float* vs = (const float*)d_in[19]; const float* vb = (const float*)d_in[20];
  const float* gamma = (const float*)d_in[21];
  float* out = (float*)d_out;

  char* ws = (char*)d_ws;
  float*     xg  = (float*)(ws);                                // 8 MB
  _Float16*  xgt = (_Float16*)(ws + 8388608);                   // 4 MB
  float*     raw = (float*)(ws + 12582912);                     // 24 MB
  _Float16*  Wr  = (_Float16*)(ws + 37748736);                  // 1.125 MB
  _Float16*  Qt  = (_Float16*)(ws + 38928384);                  // 4 MB
  _Float16*  Kt  = (_Float16*)(ws + 43122688);                  // 4 MB
  _Float16*  Vc  = (_Float16*)(ws + 47316992);                  // 4 MB

  k_prep_w<<<(3 * CCH * KTOT + 255) / 256, 256, 0, stream>>>(wq, wk, wv, Wr);

  k_split_gate<<<(BSZ * CCH * NPOS + 255) / 256, 256, 0, stream>>>(
      x, w1, b1, w2, b2, w3, b3, w4, b4, xg, xgt);

  k_qkv_gemm<<<dim3(NPOS / 16, CCH / 16, 3 * BSZ), 32, 0, stream>>>(xgt, Wr, raw);

  k_ln<<<(3 * BSZ * NPOS + 255) / 256, 256, 0, stream>>>(
      raw, bq, bk, bv, qs, qb, ks, kb, vs, vb, Qt, Kt, Vc);

  k_attn<<<BSZ * (NPOS / 16) / 8, 256, 0, stream>>>(Qt, Kt, Vc, xg, gamma, out);
}
